// CATREModel_28097676050885
// MI455X (gfx1250) — compile-verified, matrix operand order checked by reading, not run on hardware
//
#include <hip/hip_runtime.h>
#include <hip/hip_bf16.h>

#define NB 2048
#define NN 128
#define EE 256
#define RR 256

typedef __bf16 bf16_t;
typedef __attribute__((ext_vector_type(16))) __bf16 v16bf;
typedef __attribute__((ext_vector_type(8)))  __bf16 v8bf;
typedef __attribute__((ext_vector_type(8)))  float  v8f;

// ---------------- shared helpers ----------------

__device__ __forceinline__ float block_sum(float v, volatile float* red, int tid) {
  red[tid] = v; __syncthreads();
  #pragma unroll
  for (int s = 128; s > 0; s >>= 1) {
    if (tid < s) red[tid] += red[tid + s];
    __syncthreads();
  }
  float r = red[0]; __syncthreads();
  return r;
}

__device__ __forceinline__ float block_max(float v, volatile float* red, int tid) {
  red[tid] = v; __syncthreads();
  #pragma unroll
  for (int s = 128; s > 0; s >>= 1) {
    if (tid < s) red[tid] = fmaxf(red[tid], red[tid + s]);
    __syncthreads();
  }
  float r = red[0]; __syncthreads();
  return r;
}

// WMMA 16-bit operand fragment loaded straight from a bf16 row-major [rc][K]
// matrix in GLOBAL memory. ISA layout: lanes 0-15 take K = k..k+7 and k+16..k+23,
// lanes 16-31 take +8.
__device__ __forceinline__ v16bf load_frag_g(const bf16_t* rowp, int k, int lane) {
  const int half = (lane & 16) ? 8 : 0;
  const bf16_t* p = rowp + k + half;
  v8bf lo = *(const v8bf*)(p);
  v8bf hi = *(const v8bf*)(p + 16);
  v16bf f;
  #pragma unroll
  for (int i = 0; i < 8; ++i) { f[i] = lo[i]; f[i + 8] = hi[i]; }
  return f;
}

// Same fragment but sourced from fp32 global memory, converted in registers.
__device__ __forceinline__ v16bf load_frag_f32(const float* rowp, int k, int lane) {
  const int half = (lane & 16) ? 8 : 0;
  const float* p = rowp + k + half;
  float4 a0 = *(const float4*)(p);
  float4 a1 = *(const float4*)(p + 4);
  float4 a2 = *(const float4*)(p + 16);
  float4 a3 = *(const float4*)(p + 20);
  v16bf f;
  f[0]  = (bf16_t)a0.x; f[1]  = (bf16_t)a0.y; f[2]  = (bf16_t)a0.z; f[3]  = (bf16_t)a0.w;
  f[4]  = (bf16_t)a1.x; f[5]  = (bf16_t)a1.y; f[6]  = (bf16_t)a1.z; f[7]  = (bf16_t)a1.w;
  f[8]  = (bf16_t)a2.x; f[9]  = (bf16_t)a2.y; f[10] = (bf16_t)a2.z; f[11] = (bf16_t)a2.w;
  f[12] = (bf16_t)a3.x; f[13] = (bf16_t)a3.y; f[14] = (bf16_t)a3.z; f[15] = (bf16_t)a3.w;
  return f;
}

// ---------------- prep kernels: weight transpose + fp32->bf16 ----------------

// dst[n][k] = (bf16) src[k][n], with N fixed at 256. grid = K blocks, 256 threads.
__global__ __launch_bounds__(256) void tcvt(const float* __restrict__ src,
                                            bf16_t* __restrict__ dst, int K) {
  const int k = blockIdx.x, n = threadIdx.x;
  dst[(long long)n * K + k] = (bf16_t)src[(long long)k * 256 + n];
}

__global__ __launch_bounds__(256) void cvt(const float* __restrict__ src,
                                           bf16_t* __restrict__ dst, int n) {
  const int i = blockIdx.x * 256 + threadIdx.x;
  if (i < n) dst[i] = (bf16_t)src[i];
}

// ---------------- generic WMMA GEMM: C[2048,256] = A[2048,K] @ Bt[256,K]^T ----
// One wave per block; wave owns a 16-row band and all 16 col-tiles.
// blockIdx.y selects a weight/output slice (batched GEMMs share A when sA==0).

__global__ __launch_bounds__(32) void gemm_bf16(
    const bf16_t* __restrict__ A, const bf16_t* __restrict__ Bt,
    float* __restrict__ C, bf16_t* __restrict__ Cb, int K,
    long long sA, long long sB, long long sC) {
  const int lane = threadIdx.x & 31;
  const int col  = lane & 15;
  const int s    = blockIdx.y;
  const int m0   = blockIdx.x * 16;
  A  += (long long)s * sA;
  Bt += (long long)s * sB;
  const long long cbase = (long long)s * sC;

  v8f acc[16];
  #pragma unroll
  for (int t = 0; t < 16; ++t) acc[t] = (v8f){0.f,0.f,0.f,0.f,0.f,0.f,0.f,0.f};

  const bf16_t* arow = A + (long long)(m0 + col) * K;
  for (int k = 0; k < K; k += 32) {
    v16bf af = load_frag_g(arow, k, lane);
    #pragma unroll
    for (int t = 0; t < 16; ++t) {
      v16bf bf_ = load_frag_g(Bt + (long long)(t * 16 + col) * K, k, lane);
      acc[t] = __builtin_amdgcn_wmma_f32_16x16x32_bf16(
          false, af, false, bf_, (short)0, acc[t], false, false);
    }
  }
  const int rowoff = (lane & 16) ? 8 : 0;
  #pragma unroll
  for (int t = 0; t < 16; ++t)
    #pragma unroll
    for (int v = 0; v < 8; ++v) {
      long long idx = cbase + (long long)(m0 + rowoff + v) * 256 + t * 16 + col;
      C[idx] = acc[t][v];
      if (Cb) Cb[idx] = (bf16_t)acc[t][v];
    }
}

// ---------------- main fused kernel: one workgroup (8 wave32) per batch row ----

__global__ __launch_bounds__(256) void catre_main(
    const float* __restrict__ e_emb,   const float* __restrict__ nb_rel,
    const float* __restrict__ delta_t, const unsigned char* __restrict__ hist_mask,
    const float* __restrict__ log_tau1, const float* __restrict__ log_tau2,
    const float* __restrict__ sharpness, const float* __restrict__ log_gamma,
    const float* __restrict__ cs_q_b,  const float* __restrict__ cs_l_w,
    const float* __restrict__ cs_l_b,  const bf16_t* __restrict__ wt,
    const float* __restrict__ qkG,     const float* __restrict__ sqG,
    bf16_t* __restrict__ featsOut,     float* __restrict__ logitsOut) {
  __shared__ float partS [8 * EE];
  __shared__ float partMx[8 * EE];
  __shared__ float partMn[8 * EE];
  __shared__ float partSq[8 * EE];
  __shared__ float partAv[8 * EE];
  __shared__ float ddec[NN], wscA[3 * NN], hval[NN], qkl[EE];
  __shared__ float attnraw[NN], attw[NN], red[EE], miscf[4];

  const int tid  = threadIdx.x;
  const int b    = blockIdx.x;
  const int lane = tid & 31;
  const int wid  = tid >> 5;
  const int col  = lane & 15;
  const int m0   = wid * 16;
  const int rbase = m0 + ((lane & 16) ? 8 : 0);

  const float* gA = nb_rel + (size_t)b * NN * RR;

  // warm the L2/WGP$ with this block's A panel (128 KB = 1024 cachelines)
  #pragma unroll
  for (int p = 0; p < 4; ++p)
    __builtin_prefetch(gA + (tid + p * 256) * 32, 0, 3);

  // ---- prep: decay, time-scale gates, mask ----
  const float gamma = __expf(log_gamma[0]);
  const float tau1  = __expf(log_tau1[0]);
  const float tau2  = __expf(log_tau2[0]) + tau1;
  const float sharp = fminf(fmaxf(sharpness[0], 0.1f), 5.0f);
  if (tid < NN) {
    float dt = delta_t[(size_t)b * NN + tid];
    ddec[tid] = __expf(-gamma * fmaxf(dt, 0.0f));
    float wf = 1.0f / (1.0f + __expf( sharp * (dt - tau1)));
    float wc = 1.0f / (1.0f + __expf(-sharp * (dt - tau2)));
    wscA[0 * NN + tid] = wf;
    wscA[2 * NN + tid] = wc;
    wscA[1 * NN + tid] = fmaxf(0.0f, 1.0f - wf - wc);
    hval[tid] = hist_mask[(size_t)b * NN + tid] ? 1.0f : 0.0f;
  }
  __syncthreads();
  if (tid == 0) { float s = 0.f; for (int n = 0; n < NN; ++n) s += hval[n]; miscf[0] = s; }

  // ---- GEMM: msgs[b] = nb_rel[b] @ rel_w; A from fp32 global (cvt in regs),
  //      B from pre-transposed bf16 rel_w in workspace. No LDS, no barriers. ----
  v8f acc[16];
  #pragma unroll
  for (int t = 0; t < 16; ++t) acc[t] = (v8f){0.f,0.f,0.f,0.f,0.f,0.f,0.f,0.f};

  const float* arow = gA + (size_t)(m0 + col) * RR;
  for (int k = 0; k < RR; k += 32) {
    v16bf af = load_frag_f32(arow, k, lane);
    #pragma unroll
    for (int t = 0; t < 16; ++t) {
      v16bf bf_ = load_frag_g(wt + (size_t)(t * 16 + col) * RR, k, lane);
      acc[t] = __builtin_amdgcn_wmma_f32_16x16x32_bf16(
          false, af, false, bf_, (short)0, acc[t], false, false);
    }
  }
  __syncthreads();

  #pragma unroll
  for (int t = 0; t < 16; ++t)
    #pragma unroll
    for (int v = 0; v < 8; ++v)
      acc[t][v] *= ddec[rbase + v];

  const float nv = miscf[0];

  // ---- three PNA time scales ----
  for (int s = 0; s < 3; ++s) {
    const float* wscS = wscA + s * NN;
    qkl[tid] = qkG[((size_t)s * NB + b) * EE + tid];   // precomputed by WMMA GEMM
    __syncthreads();

    // pass 1: masked stats per e + raw attention logits per n
    float pat[8];
    #pragma unroll
    for (int v = 0; v < 8; ++v) pat[v] = 0.0f;
    #pragma unroll
    for (int t = 0; t < 16; ++t) {
      const int e = t * 16 + col;
      const float qke = qkl[e];
      float vsum = 0.0f, vmax = -3.4e38f, vmin = 3.4e38f, vsq = 0.0f;
      #pragma unroll
      for (int v = 0; v < 8; ++v) {
        const int n = rbase + v;
        float m = acc[t][v] * wscS[n];
        float val = hval[n];
        float mv = m * val;
        vsum += mv;
        vmax = fmaxf(vmax, mv);
        vmin = fminf(vmin, (val > 0.5f) ? m : 10000.0f);
        float c = fminf(fmaxf(mv, -10.0f), 10.0f);
        vsq += c * c;
        pat[v] += m * qke;
      }
      vsum += __shfl_xor(vsum, 16, 32);
      vmax = fmaxf(vmax, __shfl_xor(vmax, 16, 32));
      vmin = fminf(vmin, __shfl_xor(vmin, 16, 32));
      vsq += __shfl_xor(vsq, 16, 32);
      if (lane < 16) {
        partS [wid * EE + e] = vsum;
        partMx[wid * EE + e] = vmax;
        partMn[wid * EE + e] = vmin;
        partSq[wid * EE + e] = vsq;
      }
    }
    #pragma unroll
    for (int v = 0; v < 8; ++v) {
      float p = pat[v];
      p += __shfl_xor(p, 1, 32);
      p += __shfl_xor(p, 2, 32);
      p += __shfl_xor(p, 4, 32);
      p += __shfl_xor(p, 8, 32);
      if (col == 0) attnraw[rbase + v] = p * 0.0625f;   // / sqrt(E)
    }
    __syncthreads();

    // masked softmax over N
    float araw;
    if (tid < NN) {
      float a = (hval[tid] > 0.5f) ? attnraw[tid] : -10000.0f;
      if (nv < 0.5f) a = 0.0f;
      araw = a;
    } else {
      araw = -3.4e38f;
    }
    float mx = block_max(araw, red, tid);
    float ex = (tid < NN) ? __expf(araw - mx) : 0.0f;
    float se = block_sum(ex, red, tid);
    if (tid < NN) attw[tid] = ex / se;
    __syncthreads();

    // pass 2: attention-pooled value per e
    #pragma unroll
    for (int t = 0; t < 16; ++t) {
      const int e = t * 16 + col;
      float pav = 0.0f;
      #pragma unroll
      for (int v = 0; v < 8; ++v) {
        const int n = rbase + v;
        pav += acc[t][v] * wscS[n] * attw[n];
      }
      pav += __shfl_xor(pav, 16, 32);
      if (lane < 16) partAv[wid * EE + e] = pav;
    }
    __syncthreads();

    // combine 8-wave partials -> feats row (bf16, GEMM-ready [s][b][1536])
    {
      float ssum = 0.0f, smax = -3.4e38f, smin = 3.4e38f, ssq = 0.0f, sav = 0.0f;
      #pragma unroll
      for (int w = 0; w < 8; ++w) {
        ssum += partS [w * EE + tid];
        smax  = fmaxf(smax, partMx[w * EE + tid]);
        smin  = fminf(smin, partMn[w * EE + tid]);
        ssq  += partSq[w * EE + tid];
        sav  += partAv[w * EE + tid];
      }
      float nvc  = fmaxf(nv, 1.0f);
      float mean = ssum / nvc;
      bf16_t* f = featsOut + ((size_t)s * NB + b) * (6 * EE);
      f[tid]          = (bf16_t)mean;
      f[EE + tid]     = (bf16_t)smax;
      f[2 * EE + tid] = (bf16_t)fminf(fmaxf(smin, -10000.0f), 10000.0f);
      f[3 * EE + tid] = (bf16_t)sqrtf(fmaxf(ssq / nvc - mean * mean, 1e-6f));
      f[4 * EE + tid] = (bf16_t)sav;
      f[5 * EE + tid] = (bf16_t)e_emb[(size_t)b * EE + tid];
    }
    __syncthreads();
  }

  // ---- per-b cross-scale logits: (sq + cs_q_b) @ cs_l_w + cs_l_b ----
  if (tid < 3) {
    float l = cs_l_b[tid];
    const float* sqb = sqG + (size_t)b * EE;
    for (int e = 0; e < EE; ++e) l += (sqb[e] + cs_q_b[e]) * cs_l_w[e * 3 + tid];
    logitsOut[b * 3 + tid] = l;
  }
}

// ---------------- epilogue: LN+GELU per scale, softmax mix, final LN ----------

__global__ __launch_bounds__(256) void catre_epi(
    const float* __restrict__ H, const float* __restrict__ logitsG,
    const float* __restrict__ sp_b, const float* __restrict__ sp_g,
    const float* __restrict__ sp_beta, const float* __restrict__ on_g,
    const float* __restrict__ on_b, float* __restrict__ out) {
  __shared__ float red[EE];
  __shared__ float sc[3 * EE];
  const int tid = threadIdx.x;
  const int b   = blockIdx.x;
  for (int s = 0; s < 3; ++s) {
    float h = H[((size_t)s * NB + b) * EE + tid] + sp_b[s * EE + tid];
    float mean = block_sum(h, red, tid) * (1.0f / EE);
    float d = h - mean;
    float var = block_sum(d * d, red, tid) * (1.0f / EE);
    float y = d * rsqrtf(var + 1e-5f) * sp_g[s * EE + tid] + sp_beta[s * EE + tid];
    sc[s * EE + tid] = 0.5f * y * (1.0f + erff(y * 0.70710678118654752f));
  }
  __syncthreads();
  float l0 = logitsG[b * 3 + 0], l1 = logitsG[b * 3 + 1], l2 = logitsG[b * 3 + 2];
  float lm = fmaxf(l0, fmaxf(l1, l2));
  float w0 = __expf(l0 - lm), w1 = __expf(l1 - lm), w2 = __expf(l2 - lm);
  float inv = 1.0f / (w0 + w1 + w2);
  float o = (w0 * sc[tid] + w1 * sc[EE + tid] + w2 * sc[2 * EE + tid]) * inv;
  float mean = block_sum(o, red, tid) * (1.0f / EE);
  float d = o - mean;
  float var = block_sum(d * d, red, tid) * (1.0f / EE);
  out[(size_t)b * EE + tid] = d * rsqrtf(var + 1e-5f) * on_g[tid] + on_b[tid];
}

// ---------------- launcher ----------------

extern "C" void kernel_launch(void* const* d_in, const int* in_sizes, int n_in,
                              void* d_out, int out_size, void* d_ws, size_t ws_size,
                              hipStream_t stream) {
  (void)in_sizes; (void)n_in; (void)out_size; (void)ws_size;
  const float* e_emb    = (const float*)d_in[0];
  const float* nb_rel   = (const float*)d_in[1];
  const float* delta_t  = (const float*)d_in[2];
  const float* r_emb    = (const float*)d_in[3];
  const unsigned char* hist_mask = (const unsigned char*)d_in[4];
  const float* rel_w    = (const float*)d_in[5];
  const float* log_tau1 = (const float*)d_in[6];
  const float* log_tau2 = (const float*)d_in[7];
  const float* sharp    = (const float*)d_in[8];
  const float* log_gam  = (const float*)d_in[9];
  const float* attn_w   = (const float*)d_in[10];
  const float* sp_w     = (const float*)d_in[11];
  const float* sp_b     = (const float*)d_in[12];
  const float* sp_g     = (const float*)d_in[13];
  const float* sp_beta  = (const float*)d_in[14];
  const float* cs_q_w   = (const float*)d_in[15];
  const float* cs_q_b   = (const float*)d_in[16];
  const float* cs_l_w   = (const float*)d_in[17];
  const float* cs_l_b   = (const float*)d_in[18];
  const float* on_g     = (const float*)d_in[19];
  const float* on_b     = (const float*)d_in[20];
  float* out = (float*)d_out;

  char* ws = (char*)d_ws;
  bf16_t* wt     = (bf16_t*)(ws + 0);           // rel_w^T   [256][256]
  bf16_t* awt    = (bf16_t*)(ws + 131072);      // attn_w^T  [3][256][256]
  bf16_t* cqt    = (bf16_t*)(ws + 524288);      // cs_q_w^T  [256][256]
  bf16_t* spwt   = (bf16_t*)(ws + 655360);      // sp_w^T    [3][256][1536]
  bf16_t* rembf  = (bf16_t*)(ws + 3014656);     // r_emb bf16 [2048][256]
  float*  qkeyf  = (float* )(ws + 4063232);     // qkey fp32  [2048][256]
  bf16_t* qkeyb  = (bf16_t*)(ws + 6160384);     // qkey bf16
  float*  qk     = (float* )(ws + 7208960);     // [3][2048][256]
  float*  sq     = (float* )(ws + 13500416);    // [2048][256]
  float*  logits = (float* )(ws + 15597568);    // [2048][3]
  bf16_t* feats  = (bf16_t*)(ws + 15622144);    // [3][2048][1536] bf16
  float*  H      = (float* )(ws + 34496512);    // [3][2048][256]

  // weight prep (one-time per call; tiny)
  tcvt<<<dim3(256), dim3(256), 0, stream>>>(rel_w, wt, 256);
  for (int s = 0; s < 3; ++s)
    tcvt<<<dim3(256), dim3(256), 0, stream>>>(attn_w + (size_t)s * 65536,
                                              awt + (size_t)s * 65536, 256);
  tcvt<<<dim3(256), dim3(256), 0, stream>>>(cs_q_w, cqt, 256);
  for (int s = 0; s < 3; ++s)
    tcvt<<<dim3(1536), dim3(256), 0, stream>>>(sp_w + (size_t)s * 1536 * 256,
                                               spwt + (size_t)s * 256 * 1536, 1536);
  cvt<<<dim3(2048), dim3(256), 0, stream>>>(r_emb, rembf, NB * RR);

  // small WMMA GEMMs: qkey = r_emb@rel_w ; qk[s] = qkey@attn_w[s] ; sq = r_emb@cs_q_w
  gemm_bf16<<<dim3(NB / 16, 1), dim3(32), 0, stream>>>(rembf, wt, qkeyf, qkeyb, 256, 0, 0, 0);
  gemm_bf16<<<dim3(NB / 16, 3), dim3(32), 0, stream>>>(qkeyb, awt, qk, nullptr, 256,
                                                       0, 65536LL, (long long)NB * EE);
  gemm_bf16<<<dim3(NB / 16, 1), dim3(32), 0, stream>>>(rembf, cqt, sq, nullptr, 256, 0, 0, 0);

  // main fused kernel: big WMMA GEMM + PNA stats + attention -> feats (bf16), logits
  catre_main<<<dim3(NB), dim3(256), 0, stream>>>(
      e_emb, nb_rel, delta_t, hist_mask, log_tau1, log_tau2, sharp, log_gam,
      cs_q_b, cs_l_w, cs_l_b, wt, qk, sq, feats, logits);

  // batched MLP on WMMA: H[s] = feats[s] @ sp_w[s]   (K = 1536)
  gemm_bf16<<<dim3(NB / 16, 3), dim3(32), 0, stream>>>(
      feats, spwt, H, nullptr, 6 * EE,
      (long long)NB * 6 * EE, (long long)EE * 6 * EE, (long long)NB * EE);

  // epilogue: LN+GELU, cross-scale softmax mix, final LN
  catre_epi<<<dim3(NB), dim3(256), 0, stream>>>(H, logits, sp_b, sp_g, sp_beta,
                                                on_g, on_b, out);
}
